// SelfAttention_77884936945889
// MI455X (gfx1250) — matrix operand order repaired and checked by reading, not compiled, not run on hardware
//
#include <hip/hip_runtime.h>
#include <math.h>

typedef __attribute__((ext_vector_type(16))) _Float16 v16h;
typedef __attribute__((ext_vector_type(2)))  _Float16 h2;
typedef __attribute__((ext_vector_type(8)))  float    v8f;

#define WMMA_F16(A, B, C) \
    __builtin_amdgcn_wmma_f32_16x16x32_f16(false, (A), false, (B), (short)0, (C), false, false)

static constexpr int Bsz = 4;
static constexpr int T   = 2048;
static constexpr int D   = 1024;
static constexpr int DQ  = 64;

// ---------------------------------------------------------------------------
// Kernel 0: transpose + convert W [D x DQ] f32  ->  WT [DQ x D] f16
// ---------------------------------------------------------------------------
__global__ void wt_kernel(const float* __restrict__ W, _Float16* __restrict__ WT) {
    int idx = blockIdx.x * 256 + threadIdx.x;
    if (idx < D * DQ) {
        int d = idx / DQ;
        int e = idx % DQ;
        WT[(size_t)e * D + d] = (_Float16)W[idx];
    }
}

// ---------------------------------------------------------------------------
// Kernel 1: fused QKV projection. One wave per 16-row tile of (b, t).
//   Q,K stored row-major [B][T][64]; V stored transposed [B][64][T].
// ---------------------------------------------------------------------------
__global__ void __launch_bounds__(32)
qkv_kernel(const float* __restrict__ x, const _Float16* __restrict__ WT,
           float* __restrict__ Q, float* __restrict__ K, float* __restrict__ VT) {
    const int tile = blockIdx.x;              // 0 .. B*T/16-1
    const int b    = tile / (T / 16);
    const int trow = (tile % (T / 16)) * 16;
    const int lane = threadIdx.x & 31;
    const int hf   = lane >> 4;               // wave half
    const int lo   = lane & 15;

    v8f accq[4] = {}, acck[4] = {}, accv[4] = {};

    const float* xr = x + ((size_t)b * T + trow + lo) * D;   // A row = lo

    for (int kc = 0; kc < D; kc += 32) {
        // ---- A fragment: x rows, f32 -> f16 ----
        v16h a;
        #pragma unroll
        for (int j = 0; j < 8; ++j) {
            int kk = kc + hf * 16 + 2 * j;
            float2 f = *(const float2*)(xr + kk);
            a[2 * j]     = (_Float16)f.x;
            a[2 * j + 1] = (_Float16)f.y;
        }
        // ---- B fragments from pre-transposed f16 weights ----
        #pragma unroll
        for (int w = 0; w < 3; ++w) {
            const _Float16* wbase = WT + (size_t)w * DQ * D;
            #pragma unroll
            for (int nt = 0; nt < 4; ++nt) {
                const _Float16* wt = wbase + (size_t)(nt * 16 + lo) * D + kc + hf * 16;
                v16h bf;
                #pragma unroll
                for (int j = 0; j < 8; ++j) {
                    h2 p = *(const h2*)(wt + 2 * j);
                    bf[2 * j]     = p.x;
                    bf[2 * j + 1] = p.y;
                }
                if      (w == 0) accq[nt] = WMMA_F16(a, bf, accq[nt]);
                else if (w == 1) acck[nt] = WMMA_F16(a, bf, acck[nt]);
                else             accv[nt] = WMMA_F16(a, bf, accv[nt]);
            }
        }
    }

    // ---- store: C/D layout -> memory (row = v + 8*hf, col = nt*16 + lo) ----
    #pragma unroll
    for (int nt = 0; nt < 4; ++nt) {
        #pragma unroll
        for (int v = 0; v < 8; ++v) {
            size_t row = (size_t)b * T + trow + v + 8 * hf;
            Q[row * DQ + nt * 16 + lo] = accq[nt][v];
            K[row * DQ + nt * 16 + lo] = acck[nt][v];
            VT[((size_t)b * DQ + nt * 16 + lo) * T + trow + v + 8 * hf] = accv[nt][v];
        }
    }
}

// ---------------------------------------------------------------------------
// Kernel 2: causal flash attention. One wave per 16-query tile.
// ---------------------------------------------------------------------------
__global__ void __launch_bounds__(32)
attn_kernel(const float* __restrict__ Q, const float* __restrict__ K,
            const float* __restrict__ VT, float* __restrict__ Out) {
    __shared__ float Pst[16 * 32];            // P tile staging, D-layout -> A-layout

    const int tile  = blockIdx.x;
    const int b     = tile / (T / 16);
    const int qbase = (tile % (T / 16)) * 16;
    const int lane  = threadIdx.x & 31;
    const int hf    = lane >> 4;
    const int lo    = lane & 15;

    // ---- Q A-fragments over d (two K=32 chunks), scale 1/sqrt(64) folded in ----
    v16h qa[2];
    const float* qr = Q + ((size_t)b * T + qbase + lo) * DQ;
    #pragma unroll
    for (int c = 0; c < 2; ++c) {
        #pragma unroll
        for (int j = 0; j < 8; ++j) {
            int kk = c * 32 + hf * 16 + 2 * j;
            float2 f = *(const float2*)(qr + kk);
            qa[c][2 * j]     = (_Float16)(f.x * 0.125f);
            qa[c][2 * j + 1] = (_Float16)(f.y * 0.125f);
        }
    }

    float mi[8], li[8];
    #pragma unroll
    for (int v = 0; v < 8; ++v) { mi[v] = -__builtin_inff(); li[v] = 0.0f; }
    v8f acc[4] = {};

    for (int kb = 0; kb < qbase + 16; kb += 32) {     // 32-key chunks (causal bound)
        v8f S[2];
        #pragma unroll
        for (int sub = 0; sub < 2; ++sub) {
            const int kt = kb + sub * 16;
            v8f s = {};
            if (kt <= qbase + 15) {                   // wave-uniform
                const float* kr = K + ((size_t)b * T + kt + lo) * DQ;
                #pragma unroll
                for (int c = 0; c < 2; ++c) {
                    v16h kf;
                    #pragma unroll
                    for (int j = 0; j < 8; ++j) {
                        int kk = c * 32 + hf * 16 + 2 * j;
                        float2 f = *(const float2*)(kr + kk);
                        kf[2 * j]     = (_Float16)f.x;
                        kf[2 * j + 1] = (_Float16)f.y;
                    }
                    s = WMMA_F16(qa[c], kf, s);
                }
            }
            // causal mask: row query = qbase + v + 8*hf, key = kt + lo
            #pragma unroll
            for (int v = 0; v < 8; ++v) {
                int qi = qbase + v + 8 * hf;
                int ki = kt + lo;
                if (ki > qi) s[v] = -__builtin_inff();
            }
            S[sub] = s;
        }

        // ---- online softmax per row (v, hf) ----
        #pragma unroll
        for (int v = 0; v < 8; ++v) {
            float rm = fmaxf(S[0][v], S[1][v]);
            #pragma unroll
            for (int off = 1; off < 16; off <<= 1)
                rm = fmaxf(rm, __shfl_xor(rm, off, 32));   // stays within 16-lane half
            float mnew = fmaxf(mi[v], rm);
            float al   = __expf(mi[v] - mnew);
            mi[v] = mnew;
            float p0 = __expf(S[0][v] - mnew);
            float p1 = __expf(S[1][v] - mnew);
            S[0][v] = p0; S[1][v] = p1;
            float rs = p0 + p1;
            #pragma unroll
            for (int off = 1; off < 16; off <<= 1)
                rs += __shfl_xor(rs, off, 32);
            li[v] = li[v] * al + rs;
            #pragma unroll
            for (int nt = 0; nt < 4; ++nt) acc[nt][v] *= al;
        }

        // ---- P: D-layout -> LDS -> A-layout (16x32 f16) ----
        #pragma unroll
        for (int sub = 0; sub < 2; ++sub)
            #pragma unroll
            for (int v = 0; v < 8; ++v)
                Pst[(v + 8 * hf) * 32 + sub * 16 + lo] = S[sub][v];
        __syncthreads();
        v16h pa;
        #pragma unroll
        for (int j = 0; j < 8; ++j) {
            int kk = hf * 16 + 2 * j;
            float2 f = *(const float2*)(&Pst[lo * 32 + kk]);
            pa[2 * j]     = (_Float16)f.x;
            pa[2 * j + 1] = (_Float16)f.y;
        }
        __syncthreads();

        // ---- O += P * V  (V^T gives contiguous key pairs per lane) ----
        #pragma unroll
        for (int nt = 0; nt < 4; ++nt) {
            const float* vr = VT + ((size_t)b * DQ + nt * 16 + lo) * T + kb + hf * 16;
            v16h vb;
            #pragma unroll
            for (int j = 0; j < 8; ++j) {
                float2 f = *(const float2*)(vr + 2 * j);
                vb[2 * j]     = (_Float16)f.x;
                vb[2 * j + 1] = (_Float16)f.y;
            }
            acc[nt] = WMMA_F16(pa, vb, acc[nt]);
        }
    }

    // ---- epilogue: normalize and store ----
    #pragma unroll
    for (int nt = 0; nt < 4; ++nt) {
        #pragma unroll
        for (int v = 0; v < 8; ++v) {
            float o = acc[nt][v] / li[v];
            Out[((size_t)b * T + qbase + v + 8 * hf) * DQ + nt * 16 + lo] = o;
        }
    }
}

// ---------------------------------------------------------------------------
extern "C" void kernel_launch(void* const* d_in, const int* in_sizes, int n_in,
                              void* d_out, int out_size, void* d_ws, size_t ws_size,
                              hipStream_t stream) {
    const float* x  = (const float*)d_in[0];
    const float* Wq = (const float*)d_in[1];
    const float* Wk = (const float*)d_in[2];
    const float* Wv = (const float*)d_in[3];
    float* out = (float*)d_out;

    const size_t nQKV = (size_t)Bsz * T * DQ;   // 524288 floats each
    float* Qws  = (float*)d_ws;
    float* Kws  = Qws + nQKV;
    float* VTws = Kws + nQKV;
    _Float16* WT = (_Float16*)(VTws + nQKV);    // 3 * DQ * D halves

    const int nW = D * DQ;
    dim3 g0((nW + 255) / 256);
    wt_kernel<<<g0, 256, 0, stream>>>(Wq, WT + 0 * (size_t)nW);
    wt_kernel<<<g0, 256, 0, stream>>>(Wk, WT + 1 * (size_t)nW);
    wt_kernel<<<g0, 256, 0, stream>>>(Wv, WT + 2 * (size_t)nW);

    const int nTiles = Bsz * T / 16;            // 512 waves
    qkv_kernel<<<nTiles, 32, 0, stream>>>(x, WT, Qws, Kws, VTws);
    attn_kernel<<<nTiles, 32, 0, stream>>>(Qws, Kws, VTws, out);
}
